// EdgeGATLayer_66468913873057
// MI455X (gfx1250) — matrix operand order repaired and checked by reading, not compiled, not run on hardware
//
#include <hip/hip_runtime.h>

// ---------------------------------------------------------------------------
// EdgeGAT layer for MI455X (gfx1250, wave32).
//   N = 50000 nodes, E = 800000 edges, IN=128, H=4, D=32 (H*D = 128).
// Pipeline:
//   K1: feat_n = X @ W^T via v_wmma_f32_16x16x32_bf16 (LDS-staged tiles)
//   K2: el/er per (node,head); init m=-inf, s=0, out=0
//   K3: edge logits + leaky-relu + segmented atomic max
//   K4: exp(e - m[dst]) + segmented atomic sum
//   K5: scatter-aggregate  out[dst] += (eexp/s[dst]) * feat_n[src]
// ---------------------------------------------------------------------------

typedef __attribute__((ext_vector_type(16))) __bf16 v16bf;
typedef __attribute__((ext_vector_type(8)))  __bf16 v8bf;
typedef __attribute__((ext_vector_type(8)))  float  v8f;

#define IN_DIM   128
#define OUT_TOT  128   // H * D
#define NHEAD    4
#define HDIM     32
#define EDIM     16

// ---------------------------------------------------------------------------
// K1: node projection GEMM, 128x128 tile per 256-thread block (8 waves).
// Each wave owns a 16-row strip and all 8 column tiles (8 accumulators).
// K = 128 -> 4 steps of K=32 -> 32 WMMAs per wave.
// ---------------------------------------------------------------------------
__global__ __launch_bounds__(256)
void gat_node_proj_wmma(const float* __restrict__ X,   // [N,128]
                        const float* __restrict__ W,   // [128,128] (out x in)
                        float* __restrict__ Y,         // [N,128]
                        int N)
{
    __shared__ alignas(16) __bf16 sA[128][136];  // padded: 272B row stride
    __shared__ alignas(16) __bf16 sB[128][136];

    const int tid  = threadIdx.x;
    const int row0 = blockIdx.x * 128;

    // Stage weights (shared by every block; L2-hot) and the A tile as bf16.
    for (int i = tid; i < 128 * 128; i += 256) {
        int r = i >> 7, c = i & 127;
        sB[r][c] = (__bf16)W[i];
        int gr = row0 + r;
        sA[r][c] = (__bf16)(gr < N ? X[(size_t)gr * IN_DIM + c] : 0.0f);
    }
    __syncthreads();

    const int lane  = tid & 31;
    const int wave  = tid >> 5;
    const int khalf = lane >> 4;     // lane half selects K sub-range
    const int l15   = lane & 15;
    const int arow  = wave * 16 + l15;   // A: one matrix row per lane

    v8f acc[8] = {};

#pragma unroll
    for (int ks = 0; ks < 4; ++ks) {
        // --- A fragment (16x32 bf16): lanes 0-15 K={0..7,16..23}+ks*32,
        //     lanes 16-31 K={8..15,24..31}+ks*32 (ISA 7.12.2).
        const int ka = ks * 32 + khalf * 8;
        v8bf alo = *(const v8bf*)&sA[arow][ka];
        v8bf ahi = *(const v8bf*)&sA[arow][ka + 16];
        v16bf afrag;
#pragma unroll
        for (int i = 0; i < 8; ++i) { afrag[i] = alo[i]; afrag[i + 8] = ahi[i]; }

        const int kb = ks * 32 + khalf * 8;
#pragma unroll
        for (int nt = 0; nt < 8; ++nt) {
            // --- B fragment (32x16 bf16): column N = lane&15, K = {0..7,16..23}+khalf*8 (same as A).
            //     B[k][n] = W[n][k] so read W row (= output channel) from LDS.
            const int bcol = nt * 16 + l15;
            v8bf blo = *(const v8bf*)&sB[bcol][kb];
            v8bf bhi = *(const v8bf*)&sB[bcol][kb + 16];
            v16bf bfrag;
#pragma unroll
            for (int i = 0; i < 8; ++i) { bfrag[i] = blo[i]; bfrag[i + 8] = bhi[i]; }

            acc[nt] = __builtin_amdgcn_wmma_f32_16x16x32_bf16(
                false, afrag, false, bfrag, (short)0, acc[nt], false, false);
        }
    }

    // D layout: VGPR r -> M = r + khalf*8, N = lane&15 (ISA 7.12.2).
#pragma unroll
    for (int nt = 0; nt < 8; ++nt) {
        const int gcol = nt * 16 + l15;
#pragma unroll
        for (int r = 0; r < 8; ++r) {
            const int grow = row0 + wave * 16 + khalf * 8 + r;
            if (grow < N) Y[(size_t)grow * OUT_TOT + gcol] = acc[nt][r];
        }
    }
}

// ---------------------------------------------------------------------------
// K2: per-(node,head) attention logits el/er; init m, s, out.
// ---------------------------------------------------------------------------
__global__ __launch_bounds__(256)
void gat_node_logits(const float* __restrict__ featn,
                     const float* __restrict__ attn_l,
                     const float* __restrict__ attn_r,
                     float* __restrict__ el, float* __restrict__ er,
                     float* __restrict__ m,  float* __restrict__ s,
                     float* __restrict__ out, int N)
{
    int idx = blockIdx.x * blockDim.x + threadIdx.x;
    if (idx >= N * NHEAD) return;
    int n = idx >> 2, h = idx & 3;

    const float* f  = featn  + (size_t)n * OUT_TOT + h * HDIM;
    const float* al = attn_l + h * HDIM;
    const float* ar = attn_r + h * HDIM;
    float sl = 0.f, sr = 0.f;
#pragma unroll
    for (int d = 0; d < HDIM; ++d) { sl += f[d] * al[d]; sr += f[d] * ar[d]; }
    el[idx] = sl;
    er[idx] = sr;
    m[idx]  = -INFINITY;
    s[idx]  = 0.f;

    float* o = out + (size_t)n * OUT_TOT + h * HDIM;
#pragma unroll
    for (int d = 0; d < HDIM; ++d) o[d] = 0.f;
}

// float atomic max via sign-trick (lowers to global_atomic_max_i32/min_u32).
__device__ __forceinline__ void atomicMaxF(float* addr, float v)
{
    if (v >= 0.f) atomicMax((int*)addr, __float_as_int(v));
    else          atomicMin((unsigned int*)addr, __float_as_uint(v));
}

// ---------------------------------------------------------------------------
// K3: per-edge logits (edge-feature dot + gathered el/er), leaky-relu,
//     segmented max into m[dst].
// ---------------------------------------------------------------------------
__global__ __launch_bounds__(256)
void gat_edge_logits(const float* __restrict__ fe,     // [E,16]
                     const int*  __restrict__ src, const int* __restrict__ dst,
                     const float* __restrict__ attn_e, // [4,16]
                     const float* __restrict__ el, const float* __restrict__ er,
                     float* __restrict__ earr,         // [E,4]
                     float* __restrict__ m, int E)
{
    int e = blockIdx.x * blockDim.x + threadIdx.x;
    if (e >= E) return;
    const int si = src[e], di = dst[e];

    float ef[EDIM];
#pragma unroll
    for (int i = 0; i < EDIM; ++i) ef[i] = fe[(size_t)e * EDIM + i];

#pragma unroll
    for (int h = 0; h < NHEAD; ++h) {
        float ee = 0.f;
#pragma unroll
        for (int i = 0; i < EDIM; ++i) ee += ef[i] * attn_e[h * EDIM + i];
        float v = el[si * NHEAD + h] + er[di * NHEAD + h] + ee;
        v = v > 0.f ? v : 0.2f * v;              // leaky_relu(0.2)
        earr[(size_t)e * NHEAD + h] = v;
        atomicMaxF(&m[di * NHEAD + h], v);
    }
}

// ---------------------------------------------------------------------------
// K4: e_exp = exp(e - m[dst]); segmented sum into s[dst].
// ---------------------------------------------------------------------------
__global__ __launch_bounds__(256)
void gat_edge_exp(const int* __restrict__ dst,
                  float* __restrict__ earr,
                  const float* __restrict__ m,
                  float* __restrict__ s, int E)
{
    int idx = blockIdx.x * blockDim.x + threadIdx.x;
    if (idx >= E * NHEAD) return;
    int e = idx >> 2, h = idx & 3;
    int di = dst[e];
    float ex = __expf(earr[idx] - m[di * NHEAD + h]);
    earr[idx] = ex;
    atomicAdd(&s[di * NHEAD + h], ex);
}

// ---------------------------------------------------------------------------
// K5: scatter-aggregate. One thread per (edge, j) with j = h*32+d.
// idx>>7 (edge) and head are wave-uniform -> readfirstlane so the per-edge
// loads and the divide scalarize; the feat_n gather stays coalesced vector
// loads and the accumulate is a native global_atomic_add_f32 (L2-resident).
// ---------------------------------------------------------------------------
__global__ __launch_bounds__(256)
void gat_scatter(const int* __restrict__ src, const int* __restrict__ dst,
                 const float* __restrict__ earr, const float* __restrict__ s,
                 const float* __restrict__ featn, float* __restrict__ out, int E)
{
    int idx = blockIdx.x * 256 + threadIdx.x;
    int total = E * OUT_TOT;                  // 102.4M < 2^31
    if (idx >= total) return;

    int e = __builtin_amdgcn_readfirstlane(idx >> 7);         // wave-uniform
    int h = __builtin_amdgcn_readfirstlane((idx >> 5) & 3);   // wave-uniform
    int j = idx & 127;

    int si = src[e], di = dst[e];
    float a = earr[e * NHEAD + h] / s[di * NHEAD + h];        // scalar per wave
    float v = a * featn[(size_t)si * OUT_TOT + j];
    atomicAdd(&out[(size_t)di * OUT_TOT + j], v);
}

// ---------------------------------------------------------------------------
extern "C" void kernel_launch(void* const* d_in, const int* in_sizes, int n_in,
                              void* d_out, int out_size, void* d_ws, size_t ws_size,
                              hipStream_t stream)
{
    const float* X   = (const float*)d_in[0];   // feats_node [N,128]
    const float* FE  = (const float*)d_in[1];   // feats_edge [E,16]
    const int*   src = (const int*)  d_in[2];
    const int*   dst = (const int*)  d_in[3];
    const float* W   = (const float*)d_in[4];   // [128,128]
    const float* AL  = (const float*)d_in[5];   // [4,32]
    const float* AR  = (const float*)d_in[6];   // [4,32]
    const float* AE  = (const float*)d_in[7];   // [4,16]
    float* out = (float*)d_out;

    const int N = in_sizes[0] / IN_DIM;
    const int E = in_sizes[2];

    // Workspace layout (floats): feat_n | el | er | m | s | e_exp  (~42 MB)
    float* ws    = (float*)d_ws;
    float* featn = ws;
    float* el    = featn + (size_t)N * OUT_TOT;
    float* er    = el    + (size_t)N * NHEAD;
    float* m     = er    + (size_t)N * NHEAD;
    float* sv    = m     + (size_t)N * NHEAD;
    float* earr  = sv    + (size_t)N * NHEAD;

    gat_node_proj_wmma<<<(N + 127) / 128, 256, 0, stream>>>(X, W, featn, N);
    gat_node_logits<<<(N * NHEAD + 255) / 256, 256, 0, stream>>>(
        featn, AL, AR, el, er, m, sv, out, N);
    gat_edge_logits<<<(E + 255) / 256, 256, 0, stream>>>(
        FE, src, dst, AE, el, er, earr, m, E);
    gat_edge_exp<<<(E * NHEAD + 255) / 256, 256, 0, stream>>>(dst, earr, m, sv, E);
    gat_scatter<<<(E * OUT_TOT + 255) / 256, 256, 0, stream>>>(
        src, dst, earr, sv, featn, out, E);
}
